// MemoryConsolidationEngine_76845554860141
// MI455X (gfx1250) — matrix operand order repaired and checked by reading, not compiled, not run on hardware
//
#include <hip/hip_runtime.h>
#include <hip/hip_bf16.h>
#include <math.h>

typedef __attribute__((ext_vector_type(16))) _Float16 v16h;
typedef __attribute__((ext_vector_type(8)))  _Float16 v8h;
typedef __attribute__((ext_vector_type(4)))  _Float16 v4h;
typedef __attribute__((ext_vector_type(8)))  float    v8f;
typedef __attribute__((ext_vector_type(4)))  float    v4f;

#define NN    1024
#define HH    128
#define PAIRS 523776   // N*(N-1)/2, divisible by 16

// Branchless erf-GELU: Abramowitz-Stegun 7.1.26 (|err| <= 1.5e-7 on erf),
// hardware v_exp_f32 + v_rcp_f32, no divergence.
__device__ __forceinline__ float gelu_erf(float x) {
    const float ax = fabsf(x) * 0.70710678118654752f;   // |x|/sqrt(2)
    const float t  = __builtin_amdgcn_rcpf(fmaf(0.3275911f, ax, 1.0f));
    float p = fmaf(1.061405429f, t, -1.453152027f);
    p = fmaf(p, t, 1.421413741f);
    p = fmaf(p, t, -0.284496736f);
    p = fmaf(p, t, 0.254829592f);
    p = p * t;
    const float e  = __expf(-ax * ax);
    const float er = copysignf(fmaf(-p, e, 1.0f), x);
    return 0.5f * x * (1.0f + er);
}
__device__ __forceinline__ float sigmoidf_(float x) {
    return __builtin_amdgcn_rcpf(1.0f + __expf(-x));
}

// ---------------------------------------------------------------------------
// Kernel 1: U = emb @ cW1[:128] + cb1 ; V = emb @ cW1[128:]
//           A = emb @ aW1[:128] + ab1 ; B = emb @ aW1[128:]
// grid 512 x 256 threads; thread -> (n, h)
// ---------------------------------------------------------------------------
__global__ __launch_bounds__(256)
void uvab_kernel(const float* __restrict__ emb,
                 const float* __restrict__ cW1, const float* __restrict__ cb1,
                 const float* __restrict__ aW1, const float* __restrict__ ab1,
                 float* __restrict__ U, float* __restrict__ V,
                 float* __restrict__ Aw, float* __restrict__ Bw) {
    __shared__ float rows[2 * HH];
    const int tid = threadIdx.x;
    const int n0  = blockIdx.x * 2;
    rows[tid] = emb[n0 * HH + tid];
    __syncthreads();
    const int nl = tid >> 7;
    const int h  = tid & 127;
    const int n  = n0 + nl;
    const float* er = &rows[nl * HH];
    float au = 0.f, av = 0.f, aa = 0.f, ab = 0.f;
#pragma unroll 4
    for (int k = 0; k < HH; ++k) {
        const float e = er[k];
        au += e * cW1[k * HH + h];
        av += e * cW1[(k + HH) * HH + h];
        aa += e * aW1[k * HH + h];
        ab += e * aW1[(k + HH) * HH + h];
    }
    U [n * HH + h] = au + cb1[h];
    V [n * HH + h] = av;
    Aw[n * HH + h] = aa + ab1[h];
    Bw[n * HH + h] = ab;
}

// ---------------------------------------------------------------------------
// Kernel 2: importance head, one block (64 threads) per trace
// ---------------------------------------------------------------------------
__global__ __launch_bounds__(64)
void imp_kernel(const float* __restrict__ emb, const float* __restrict__ tf,
                const float* __restrict__ iW1, const float* __restrict__ ib1,
                const float* __restrict__ iW2, const float* __restrict__ ib2,
                float* __restrict__ outI) {
    __shared__ float fin[133];
    __shared__ float red[64];
    const int n = blockIdx.x;
    const int t = threadIdx.x;           // 0..63
    fin[t]      = emb[n * HH + t];
    fin[t + 64] = emb[n * HH + 64 + t];
    if (t < 5) fin[128 + t] = tf[n * 5 + t];
    __syncthreads();
    float acc = ib1[t];
#pragma unroll 7
    for (int k = 0; k < 133; ++k) acc += fin[k] * iW1[k * 64 + t];
    red[t] = gelu_erf(acc) * iW2[t];
    __syncthreads();
    for (int s = 32; s > 0; s >>= 1) {
        if (t < s) red[t] += red[t + s];
        __syncthreads();
    }
    if (t == 0) outI[n] = sigmoidf_(red[0] + ib2[0]);
}

// ---------------------------------------------------------------------------
// Pair index -> (i,j) for triu(k=1) row-major order. C(i)=i*(2N-1-i)/2.
// Called once per 16-pair tile; iterated incrementally inside.
// ---------------------------------------------------------------------------
__device__ __forceinline__ void pair_ij(int p, int& i, int& j) {
    const double disc = 2047.0 * 2047.0 - 8.0 * (double)p;
    int ii = (int)((2047.0 - sqrt(disc)) * 0.5);
    if (ii < 0) ii = 0;
    while ((long long)(ii + 1) * (2047 - (ii + 1)) / 2 <= (long long)p) ++ii;
    while ((long long)ii * (2047 - ii) / 2 > (long long)p) --ii;
    const int ci = (int)((long long)ii * (2047 - ii) / 2);
    i = ii;
    j = ii + 1 + (p - ci);
}

__device__ __forceinline__ v16h combine16(v8h lo, v8h hi) {
    return __builtin_shufflevector(lo, hi, 0,1,2,3,4,5,6,7,8,9,10,11,12,13,14,15);
}

// ---------------------------------------------------------------------------
// Kernel 3: pair MLPs via WMMA. 128 threads = 4 waves; each wave does 4
// 16-pair tiles. grid = 32736/16 = 2046 blocks.
// ---------------------------------------------------------------------------
__global__ __launch_bounds__(128)
void pair_kernel(const float* __restrict__ U,   const float* __restrict__ V,
                 const float* __restrict__ Aw,  const float* __restrict__ Bw,
                 const float* __restrict__ ln_g, const float* __restrict__ ln_b,
                 const float* __restrict__ cW2, const float* __restrict__ cb2,
                 const float* __restrict__ cW3, const float* __restrict__ cb3,
                 const float* __restrict__ aW2, const float* __restrict__ ab2,
                 const float* __restrict__ aW3, const float* __restrict__ ab3,
                 float* __restrict__ out) {
    // packed WMMA-B layout weights + per-wave hidden tiles
    __shared__ __align__(16) _Float16 sWc[16 * 512];       // cons: 4 kc x 4 nt
    __shared__ __align__(16) _Float16 sWa[8 * 512];        // assoc: 4 kc x 2 nt
    __shared__ __align__(16) _Float16 sHc[4][16][HH];
    __shared__ __align__(16) _Float16 sHa[4][16][HH];
    __shared__ float sLng[HH], sLnb[HH], sC3[64], sCb2[64], sA3[32], sAb2[32];

    const int tid = threadIdx.x;

    // ---- init: swizzle layer-2 weights into WMMA-B layout (lane=col, half=K)
    for (int idx = tid; idx < 16 * 512; idx += 128) {
        const int b = idx >> 9, l = (idx >> 4) & 31, e = idx & 15;
        const int kc = b >> 2, nt = b & 3;
        const int col = l & 15, kbase = (l >> 4) * 16;
        const int k = kc * 32 + kbase + e;
        sWc[idx] = (_Float16)cW2[k * 64 + nt * 16 + col];
    }
    for (int idx = tid; idx < 8 * 512; idx += 128) {
        const int b = idx >> 9, l = (idx >> 4) & 31, e = idx & 15;
        const int kc = b >> 1, nt = b & 1;
        const int col = l & 15, kbase = (l >> 4) * 16;
        const int k = kc * 32 + kbase + e;
        sWa[idx] = (_Float16)aW2[k * 32 + nt * 16 + col];
    }
    sLng[tid] = ln_g[tid];
    sLnb[tid] = ln_b[tid];
    if (tid < 64) { sC3[tid] = cW3[tid]; sCb2[tid] = cb2[tid]; }
    if (tid < 32) { sA3[tid] = aW3[tid]; sAb2[tid] = ab2[tid]; }
    __syncthreads();

    const int wave = tid >> 5;
    const int lane = tid & 31;
    const float cb3v = cb3[0], ab3v = ab3[0];
    const int col  = lane & 15;
    const int arow = lane & 15;
    const int koff = (lane >> 4) * 16;

    for (int it = 0; it < 4; ++it) {
        const int t = blockIdx.x * 16 + wave * 4 + it;
        const int pbase = t * 16;

        // ---------------- phase 1: build f16 hidden rows in LDS ------------
        const int f0 = lane * 4;
        int pi, pj;
        pair_ij(pbase, pi, pj);                 // one f64 sqrt per tile
        for (int r = 0; r < 16; ++r) {
            const v4f xu = *(const v4f*)(U + (size_t)pi * HH + f0);
            const v4f xv = *(const v4f*)(V + (size_t)pj * HH + f0);
            const v4f x  = xu + xv;                       // pre-LN (cb1 folded)
            float s  = x[0] + x[1] + x[2] + x[3];
            float ss = x[0]*x[0] + x[1]*x[1] + x[2]*x[2] + x[3]*x[3];
            for (int m = 16; m >= 1; m >>= 1) {
                s  += __shfl_xor(s,  m, 32);
                ss += __shfl_xor(ss, m, 32);
            }
            const float mean = s * (1.0f / 128.0f);
            const float var  = ss * (1.0f / 128.0f) - mean * mean;
            const float rstd = rsqrtf(var + 1e-5f);
            v4h hc;
#pragma unroll
            for (int c = 0; c < 4; ++c) {
                const float y = (x[c] - mean) * rstd * sLng[f0 + c] + sLnb[f0 + c];
                hc[c] = (_Float16)gelu_erf(y);
            }
            *(v4h*)&sHc[wave][r][f0] = hc;

            const v4f au = *(const v4f*)(Aw + (size_t)pi * HH + f0);
            const v4f av = *(const v4f*)(Bw + (size_t)pj * HH + f0);
            const v4f ax = au + av;                       // ab1 folded
            v4h ha;
#pragma unroll
            for (int c = 0; c < 4; ++c) {
                const float y = ax[c] > 0.0f ? ax[c] : 0.0f;
                ha[c] = (_Float16)y;
            }
            *(v4h*)&sHa[wave][r][f0] = ha;

            if (++pj == NN) { ++pi; pj = pi + 1; }   // advance to next pair
        }

        // ---------------- phase 2: consolidation (128 -> 64 -> 1) ----------
        v16h afc[4];
#pragma unroll
        for (int kc = 0; kc < 4; ++kc) {
            const v8h lo = *(const v8h*)&sHc[wave][arow][kc * 32 + koff];
            const v8h hi = *(const v8h*)&sHc[wave][arow][kc * 32 + 8 + koff];
            afc[kc] = combine16(lo, hi);
        }
        float part[8] = {0.f, 0.f, 0.f, 0.f, 0.f, 0.f, 0.f, 0.f};
#pragma unroll
        for (int nt = 0; nt < 4; ++nt) {
            v8f acc = {};
#pragma unroll
            for (int kc = 0; kc < 4; ++kc) {
                const _Float16* bp = &sWc[(kc * 4 + nt) * 512 + lane * 16];
                const v16h bf = combine16(*(const v8h*)bp, *(const v8h*)(bp + 8));
                acc = __builtin_amdgcn_wmma_f32_16x16x32_f16(
                    false, afc[kc], false, bf, (short)0, acc, false, false);
            }
            const float bias = sCb2[nt * 16 + col];
            const float w3   = sC3[nt * 16 + col];
#pragma unroll
            for (int v = 0; v < 8; ++v)
                part[v] += gelu_erf(acc[v] + bias) * w3;
        }
#pragma unroll
        for (int v = 0; v < 8; ++v)
            for (int m = 8; m >= 1; m >>= 1)
                part[v] += __shfl_xor(part[v], m, 16);
        if ((lane & 15) == 0) {
            const int rbase = (lane >> 4) * 8;
#pragma unroll
            for (int v = 0; v < 8; ++v)
                out[pbase + rbase + v] = sigmoidf_(part[v] + cb3v);
        }

        // ---------------- phase 3: association (128 -> 32 -> 1) ------------
        v16h afa[4];
#pragma unroll
        for (int kc = 0; kc < 4; ++kc) {
            const v8h lo = *(const v8h*)&sHa[wave][arow][kc * 32 + koff];
            const v8h hi = *(const v8h*)&sHa[wave][arow][kc * 32 + 8 + koff];
            afa[kc] = combine16(lo, hi);
        }
        float parta[8] = {0.f, 0.f, 0.f, 0.f, 0.f, 0.f, 0.f, 0.f};
#pragma unroll
        for (int nt = 0; nt < 2; ++nt) {
            v8f acc = {};
#pragma unroll
            for (int kc = 0; kc < 4; ++kc) {
                const _Float16* bp = &sWa[(kc * 2 + nt) * 512 + lane * 16];
                const v16h bf = combine16(*(const v8h*)bp, *(const v8h*)(bp + 8));
                acc = __builtin_amdgcn_wmma_f32_16x16x32_f16(
                    false, afa[kc], false, bf, (short)0, acc, false, false);
            }
            const float bias = sAb2[nt * 16 + col];
            const float w3   = sA3[nt * 16 + col];
#pragma unroll
            for (int v = 0; v < 8; ++v) {
                const float y = acc[v] + bias;
                parta[v] += (y > 0.0f ? y : 0.0f) * w3;
            }
        }
#pragma unroll
        for (int v = 0; v < 8; ++v)
            for (int m = 8; m >= 1; m >>= 1)
                parta[v] += __shfl_xor(parta[v], m, 16);
        if ((lane & 15) == 0) {
            const int rbase = (lane >> 4) * 8;
#pragma unroll
            for (int v = 0; v < 8; ++v)
                out[PAIRS + pbase + rbase + v] = sigmoidf_(parta[v] + ab3v);
        }
    }
}

// ---------------------------------------------------------------------------
extern "C" void kernel_launch(void* const* d_in, const int* in_sizes, int n_in,
                              void* d_out, int out_size, void* d_ws, size_t ws_size,
                              hipStream_t stream) {
    const float* emb   = (const float*)d_in[0];
    const float* trace = (const float*)d_in[1];
    const float* cW1 = (const float*)d_in[2];  const float* cb1 = (const float*)d_in[3];
    const float* lng = (const float*)d_in[4];  const float* lnb = (const float*)d_in[5];
    const float* cW2 = (const float*)d_in[6];  const float* cb2 = (const float*)d_in[7];
    const float* cW3 = (const float*)d_in[8];  const float* cb3 = (const float*)d_in[9];
    const float* aW1 = (const float*)d_in[10]; const float* ab1 = (const float*)d_in[11];
    const float* aW2 = (const float*)d_in[12]; const float* ab2 = (const float*)d_in[13];
    const float* aW3 = (const float*)d_in[14]; const float* ab3 = (const float*)d_in[15];
    const float* iW1 = (const float*)d_in[16]; const float* ib1 = (const float*)d_in[17];
    const float* iW2 = (const float*)d_in[18]; const float* ib2 = (const float*)d_in[19];
    float* out = (float*)d_out;

    float* U  = (float*)d_ws;          // 1024*128 each
    float* V  = U  + NN * HH;
    float* Ap = V  + NN * HH;
    float* Bp = Ap + NN * HH;

    uvab_kernel<<<512, 256, 0, stream>>>(emb, cW1, cb1, aW1, ab1, U, V, Ap, Bp);
    imp_kernel<<<NN, 64, 0, stream>>>(emb, trace, iW1, ib1, iW2, ib2, out + 2 * PAIRS);
    pair_kernel<<<PAIRS / 16 / 16, 128, 0, stream>>>(U, V, Ap, Bp, lng, lnb,
                                                     cW2, cb2, cW3, cb3,
                                                     aW2, ab2, aW3, ab3, out);
}
